// PABlock_3882650436003
// MI455X (gfx1250) — matrix operand order repaired and checked by reading, not compiled, not run on hardware
//
#include <hip/hip_runtime.h>
#include <stdint.h>

// ---------- CDNA5 WMMA types ----------
typedef __bf16 bf16_t;
typedef bf16_t v16bf __attribute__((ext_vector_type(16)));
typedef float  v8f   __attribute__((ext_vector_type(8)));

union BV { v16bf v; uint4 q[2]; };

__device__ __forceinline__ v8f wmma_bf16(v16bf a, v16bf b, v8f c) {
  return __builtin_amdgcn_wmma_f32_16x16x32_bf16(false, a, false, b, (short)0, c,
                                                 false, false);
}

__device__ __forceinline__ uint16_t f2bf(float f) {
  uint32_t u = __float_as_uint(f);
  uint32_t r = (u + 0x7FFFu + ((u >> 16) & 1u)) >> 16;  // RNE
  return (uint16_t)r;
}

// Problem constants
#define BB 4
#define CC_ 512
#define NN_ 4096
#define CQ 64

// ---------- convert kernels ----------
__global__ void k_f32_to_bf16(const float* __restrict__ src,
                              uint16_t* __restrict__ dst, int n) {
  int i = blockIdx.x * blockDim.x + threadIdx.x;
  if (i < n) dst[i] = f2bf(src[i]);
}

// x:[B,C,N] fp32 -> xT:[B,N,C] bf16
__global__ void k_x_to_xT(const float* __restrict__ x, uint16_t* __restrict__ xT) {
  int i = blockIdx.x * blockDim.x + threadIdx.x;   // B*C*N threads
  int n = i & (NN_ - 1);
  int c = (i >> 12) & (CC_ - 1);
  int b = i >> 21;
  xT[(((size_t)b << 12) + n) * CC_ + c] = f2bf(x[i]);
}

// ---------- projection GEMM: out = W[Co,C] * x[C,N] + bias ----------
// Each wave computes a 16(M) x 64(N) strip, K=512 in steps of 32.
// outTransposed=1: out[b, n, m] (for Q,K, Co=64); 0: out[b, m, n] (V, Co=512)
__global__ __launch_bounds__(256) void k_proj_gemm(
    const uint16_t* __restrict__ Wb, const float* __restrict__ bias,
    const uint16_t* __restrict__ xT, uint16_t* __restrict__ outp,
    int Co, int outTransposed) {
  const int lane = threadIdx.x & 31;
  const int wave = threadIdx.x >> 5;
  const int lo = lane & 15;
  const int hi = lane >> 4;

  const int strip = blockIdx.x * 8 + wave;
  const int stripsPerBatch = (Co >> 4) * 64;
  const int b  = strip / stripsPerBatch;
  const int r0 = strip % stripsPerBatch;
  const int m0 = (r0 >> 6) << 4;
  const int n0 = (r0 & 63) << 6;

  const v8f z8 = {0.f, 0.f, 0.f, 0.f, 0.f, 0.f, 0.f, 0.f};
  v8f acc[4] = {z8, z8, z8, z8};

  // A-layout lane addressing (16-bit A 16x32): lanes<16 K 0..7 & 16..23, etc.
  const uint16_t* Arow  = Wb + (size_t)(m0 + lo) * CC_ + (hi ? 8 : 0);
  // B-layout (16-bit B 32x16): lane = column n, K 0..7 & 16..23 (lanes<16), etc.
  const uint16_t* Bbase = xT + ((size_t)b * NN_ + n0 + lo) * CC_ + (hi ? 8 : 0);

  for (int kk = 0; kk < CC_; kk += 32) {
    BV a;
    a.q[0] = *(const uint4*)(Arow + kk);
    a.q[1] = *(const uint4*)(Arow + kk + 16);
#pragma unroll
    for (int nn = 0; nn < 4; ++nn) {
      BV bm;
      const uint16_t* p = Bbase + (size_t)nn * 16 * CC_ + kk;
      bm.q[0] = *(const uint4*)(p);
      bm.q[1] = *(const uint4*)(p + 16);
      acc[nn] = wmma_bf16(a.v, bm.v, acc[nn]);
    }
  }

  float bvv[8];
#pragma unroll
  for (int r = 0; r < 8; ++r) bvv[r] = bias[m0 + r + 8 * hi];

#pragma unroll
  for (int nn = 0; nn < 4; ++nn) {
#pragma unroll
    for (int r = 0; r < 8; ++r) {
      const int m = m0 + r + 8 * hi;           // C/D layout: M = r + 8*(lane/16)
      const int n = n0 + nn * 16 + lo;         //             N = lane%16
      const uint16_t val = f2bf(acc[nn][r] + bvv[r]);
      if (outTransposed) outp[((size_t)b * NN_ + n) * Co + m] = val;
      else               outp[((size_t)b * Co + m) * NN_ + n] = val;
    }
  }
}

// ---------- flash attention (cooperative S, TJ = 128) ----------
// block = 8 waves handling (batch b, 32 queries). Per 128-j step the 16
// (i-tile t, j-subtile jj) S tiles are computed exactly once: wave w owns
// i-tile t = w>>2 and the j-subtile pair {2(w&3), 2(w&3)+1}. Softmax stats
// flow through LDS partials; P (32 x 128, bf16) is shared in LDS.
// For the PV GEMM, wave w owns output channels [64w, 64w+64).
struct AttnBuf {
  uint16_t P[2][16 * 128];  // [t][row*128 + j] bf16 probability tiles
  float    M[2][16 * 8];    // [t][row*8 + jj] partial row maxima
  float    S[2][16 * 8];    // [t][row*8 + jj] partial row sums
};

__global__ __launch_bounds__(256) void k_attn(
    const uint16_t* __restrict__ Qb,   // [B, N, 64] bf16 (q^T)
    const uint16_t* __restrict__ Kb,   // [B, N, 64] bf16 (k^T)
    const uint16_t* __restrict__ Vb,   // [B, C, N]  bf16
    const float* __restrict__ x,       // [B, C, N]  fp32
    const float* __restrict__ gamma,
    float* __restrict__ outp) {
  __shared__ union ShBuf {
    AttnBuf ab[2];               // double buffered: 2 * 10240 B
    float   stage[8][64 * 17];   // epilogue transpose staging (34816 B)
  } sh;

  const int lane = threadIdx.x & 31;
  const int wave = threadIdx.x >> 5;
  const int lo = lane & 15;
  const int hi = lane >> 4;
  const int b  = blockIdx.x >> 7;            // 128 query-blocks per batch
  const int i0 = (blockIdx.x & 127) << 5;    // 32 queries per block
  const int cbase = wave << 6;
  const int t_own  = wave >> 2;              // this wave's S i-tile
  const int jjp    = (wave & 3) * 2;         // first of its j-subtile pair

  // Q A-tiles for the wave's own i-tile (K = 64 -> 2 WMMA K-steps).
  BV qa[2];
  {
    const uint16_t* qp =
        Qb + ((size_t)b * NN_ + i0 + t_own * 16 + lo) * CQ + (hi ? 8 : 0);
    qa[0].q[0] = *(const uint4*)(qp);
    qa[0].q[1] = *(const uint4*)(qp + 16);
    qa[1].q[0] = *(const uint4*)(qp + 32);
    qa[1].q[1] = *(const uint4*)(qp + 48);
  }

  const v8f z8 = {0.f, 0.f, 0.f, 0.f, 0.f, 0.f, 0.f, 0.f};
  v8f acc[2][4];
#pragma unroll
  for (int t = 0; t < 2; ++t)
#pragma unroll
    for (int cc = 0; cc < 4; ++cc) acc[t][cc] = z8;

  float mrow[2][8], lrow[2][8];
#pragma unroll
  for (int t = 0; t < 2; ++t)
#pragma unroll
    for (int r = 0; r < 8; ++r) { mrow[t][r] = -3.0e38f; lrow[t][r] = 0.f; }

  const uint16_t* Kbase = Kb + ((size_t)b * NN_ + lo) * CQ + (hi ? 8 : 0);
  const uint16_t* Vbase =
      Vb + ((size_t)b * CC_ + cbase + lo) * NN_ + (hi ? 8 : 0);

  for (int j0 = 0; j0 < NN_; j0 += 128) {
    AttnBuf* B_ = &sh.ab[(j0 >> 7) & 1];

    // ---- phase 1: this wave's two 16x16 S tiles + partial row maxima ----
    v8f S8[2] = {z8, z8};
    float mc[2][8];
#pragma unroll
    for (int e = 0; e < 2; ++e) {
      const uint16_t* p0 = Kbase + (size_t)(j0 + (jjp + e) * 16) * CQ;
      BV kb;
      kb.q[0] = *(const uint4*)(p0);
      kb.q[1] = *(const uint4*)(p0 + 16);
      S8[e] = wmma_bf16(qa[0].v, kb.v, S8[e]);
      kb.q[0] = *(const uint4*)(p0 + 32);
      kb.q[1] = *(const uint4*)(p0 + 48);
      S8[e] = wmma_bf16(qa[1].v, kb.v, S8[e]);
#pragma unroll
      for (int r = 0; r < 8; ++r) {
        float m = S8[e][r];
        m = fmaxf(m, __shfl_xor(m, 1, 16));
        m = fmaxf(m, __shfl_xor(m, 2, 16));
        m = fmaxf(m, __shfl_xor(m, 4, 16));
        m = fmaxf(m, __shfl_xor(m, 8, 16));
        mc[e][r] = m;
      }
    }
    if (lo == 0) {
#pragma unroll
      for (int e = 0; e < 2; ++e)
#pragma unroll
        for (int r = 0; r < 8; ++r)
          B_->M[t_own][(r + 8 * hi) * 8 + jjp + e] = mc[e][r];
    }
    __syncthreads();

    // ---- phase 2: combine maxima (all waves, registers), own tiles -> P ----
    float scale[2][8];
#pragma unroll
    for (int t = 0; t < 2; ++t) {
#pragma unroll
      for (int r = 0; r < 8; ++r) {
        const float4 m0v = *(const float4*)&B_->M[t][(r + 8 * hi) * 8];
        const float4 m1v = *(const float4*)&B_->M[t][(r + 8 * hi) * 8 + 4];
        const float m8 = fmaxf(fmaxf(fmaxf(m0v.x, m0v.y), fmaxf(m0v.z, m0v.w)),
                               fmaxf(fmaxf(m1v.x, m1v.y), fmaxf(m1v.z, m1v.w)));
        const float mn = fmaxf(mrow[t][r], m8);
        scale[t][r] = __expf(mrow[t][r] - mn);
        mrow[t][r] = mn;
      }
    }
#pragma unroll
    for (int e = 0; e < 2; ++e) {
#pragma unroll
      for (int r = 0; r < 8; ++r) {
        float p = __expf(S8[e][r] - mrow[t_own][r]);
        B_->P[t_own][(r + 8 * hi) * 128 + (jjp + e) * 16 + lo] = f2bf(p);
        p += __shfl_xor(p, 1, 16);
        p += __shfl_xor(p, 2, 16);
        p += __shfl_xor(p, 4, 16);
        p += __shfl_xor(p, 8, 16);
        mc[e][r] = p;  // reuse as partial row sum
      }
    }
    if (lo == 0) {
#pragma unroll
      for (int e = 0; e < 2; ++e)
#pragma unroll
        for (int r = 0; r < 8; ++r)
          B_->S[t_own][(r + 8 * hi) * 8 + jjp + e] = mc[e][r];
    }
    __syncthreads();

    // ---- phase 3: fold sums, rescale acc, PV GEMM (K = 128) from shared P ----
#pragma unroll
    for (int t = 0; t < 2; ++t) {
#pragma unroll
      for (int r = 0; r < 8; ++r) {
        const float4 s0v = *(const float4*)&B_->S[t][(r + 8 * hi) * 8];
        const float4 s1v = *(const float4*)&B_->S[t][(r + 8 * hi) * 8 + 4];
        lrow[t][r] = lrow[t][r] * scale[t][r] +
                     (s0v.x + s0v.y + s0v.z + s0v.w) +
                     (s1v.x + s1v.y + s1v.z + s1v.w);
      }
#pragma unroll
      for (int cc = 0; cc < 4; ++cc)
#pragma unroll
        for (int r = 0; r < 8; ++r) acc[t][cc][r] *= scale[t][r];
    }
    const uint16_t* Pb0 = &B_->P[0][0];
#pragma unroll
    for (int ks = 0; ks < 4; ++ks) {
      BV pa0, pa1;
      const uint16_t* pr = Pb0 + lo * 128 + ks * 32 + (hi ? 8 : 0);
      pa0.q[0] = *(const uint4*)pr;
      pa0.q[1] = *(const uint4*)(pr + 16);
      pa1.q[0] = *(const uint4*)(pr + 2048);
      pa1.q[1] = *(const uint4*)(pr + 2048 + 16);
#pragma unroll
      for (int cc = 0; cc < 4; ++cc) {
        BV vbm;
        const uint16_t* vp = Vbase + (size_t)cc * 16 * NN_ + j0 + ks * 32;
        vbm.q[0] = *(const uint4*)vp;
        vbm.q[1] = *(const uint4*)(vp + 16);
        acc[0][cc] = wmma_bf16(pa0.v, vbm.v, acc[0][cc]);
        acc[1][cc] = wmma_bf16(pa1.v, vbm.v, acc[1][cc]);
      }
    }
    if (j0 + 128 < NN_) {  // pull next K/V tiles toward L2 (global_prefetch_b8)
      __builtin_prefetch(Kbase + (size_t)(j0 + 128 + jjp * 16) * CQ, 0, 0);
      __builtin_prefetch(Vbase + (size_t)(j0 + 128), 0, 0);
    }
  }

  // ---- epilogue: divide by softmax sum, transpose via LDS, fused residual ----
  __syncthreads();  // retire all shared-buffer usage before stage[] aliases it
  const float g = gamma[0];
#pragma unroll
  for (int t = 0; t < 2; ++t) {
#pragma unroll
    for (int cc = 0; cc < 4; ++cc)
#pragma unroll
      for (int r = 0; r < 8; ++r)
        sh.stage[wave][(cc * 16 + lo) * 17 + r + 8 * hi] =
            acc[t][cc][r] / lrow[t][r];
#pragma unroll 4
    for (int cp = 0; cp < 32; ++cp) {
      const int cl = cp * 2 + hi;
      const float v = sh.stage[wave][cl * 17 + lo];
      const size_t gi =
          ((size_t)b * CC_ + cbase + cl) * NN_ + i0 + t * 16 + lo;
      outp[gi] = g * v + x[gi];
    }
  }
}

// ---------- host launcher ----------
extern "C" void kernel_launch(void* const* d_in, const int* in_sizes, int n_in,
                              void* d_out, int out_size, void* d_ws, size_t ws_size,
                              hipStream_t stream) {
  const float* x     = (const float*)d_in[0];
  const float* Wq    = (const float*)d_in[1];
  const float* bq    = (const float*)d_in[2];
  const float* Wk    = (const float*)d_in[3];
  const float* bk    = (const float*)d_in[4];
  const float* Wv    = (const float*)d_in[5];
  const float* bv    = (const float*)d_in[6];
  const float* gamma = (const float*)d_in[7];
  float* out = (float*)d_out;

  char* ws = (char*)d_ws;
  size_t off = 0;
  auto alloc = [&](size_t bytes) {
    char* p = ws + off;
    off = (off + bytes + 255) & ~(size_t)255;
    return p;
  };
  uint16_t* xT  = (uint16_t*)alloc((size_t)BB * NN_ * CC_ * 2);
  uint16_t* Wqb = (uint16_t*)alloc((size_t)CQ * CC_ * 2);
  uint16_t* Wkb = (uint16_t*)alloc((size_t)CQ * CC_ * 2);
  uint16_t* Wvb = (uint16_t*)alloc((size_t)CC_ * CC_ * 2);
  uint16_t* Qb  = (uint16_t*)alloc((size_t)BB * NN_ * CQ * 2);
  uint16_t* Kb  = (uint16_t*)alloc((size_t)BB * NN_ * CQ * 2);
  uint16_t* Vb  = (uint16_t*)alloc((size_t)BB * CC_ * NN_ * 2);
  (void)ws_size; (void)in_sizes; (void)n_in; (void)out_size;

  k_f32_to_bf16<<<(CQ * CC_ + 255) / 256, 256, 0, stream>>>(Wq, Wqb, CQ * CC_);
  k_f32_to_bf16<<<(CQ * CC_ + 255) / 256, 256, 0, stream>>>(Wk, Wkb, CQ * CC_);
  k_f32_to_bf16<<<(CC_ * CC_ + 255) / 256, 256, 0, stream>>>(Wv, Wvb, CC_ * CC_);
  k_x_to_xT<<<(BB * CC_ * NN_) / 256, 256, 0, stream>>>(x, xT);

  // strips = B * (Co/16) * 64 ; 8 wave-strips per block
  k_proj_gemm<<<(BB * (CQ / 16) * 64) / 8, 256, 0, stream>>>(Wqb, bq, xT, Qb, CQ, 1);
  k_proj_gemm<<<(BB * (CQ / 16) * 64) / 8, 256, 0, stream>>>(Wkb, bk, xT, Kb, CQ, 1);
  k_proj_gemm<<<(BB * (CC_ / 16) * 64) / 8, 256, 0, stream>>>(Wvb, bv, xT, Vb, CC_, 0);

  k_attn<<<BB * (NN_ / 32), 256, 0, stream>>>(Qb, Kb, Vb, x, gamma, out);
}
